// Neural_Expert_SDE_integrator_29652454212140
// MI455X (gfx1250) — matrix operand order repaired and checked by reading, not compiled, not run on hardware
//
#include <hip/hip_runtime.h>
#include <hip/hip_bf16.h>

// ---------------------------------------------------------------------------
// CDNA5 (gfx1250) WMMA types
// ---------------------------------------------------------------------------
typedef __bf16 v16bf __attribute__((ext_vector_type(16)));
typedef float  v8f   __attribute__((ext_vector_type(8)));

union FragAB {
    v16bf v;
    uint4 q[2];
};

#define N_TOTAL   8192      // B*S rows
#define L_DIM     100
#define AUG       102       // y(100) + logqp + T
#define T_STEPS   40
#define KPAD      416       // 400 padded to 13*32
#define TM        64        // rows per block
#define SIGMA_C   0.5f

// frag buffer metadata (elements, bf16)
#define FRAG_TOTAL 763392
#define FRAG_BYTES 1526784

__device__ __forceinline__ unsigned short f2bf(float f) {
    unsigned int u = __float_as_uint(f);
    unsigned int r = u + 0x7FFFu + ((u >> 16) & 1u);   // round-to-nearest-even
    return (unsigned short)(r >> 16);
}

// ---------------------------------------------------------------------------
// Kernel 1: repack drift weights (fp32 row-major K x C) into bf16 WMMA
// B-fragment layout.  Fragment (ctile,kstep) = 32 lanes x 16 elems, lane's 16
// elements contiguous.  B element (lane,j): c = ctile*16+(lane&15),
// k = kstep*32 + ((lane>>4)<<4) + j.   Zero-padded outside [K)x[C).
// ---------------------------------------------------------------------------
__global__ void prep_frags(const float* __restrict__ W0, const float* __restrict__ W1,
                           const float* __restrict__ W2, const float* __restrict__ W3,
                           const float* __restrict__ W4, const float* __restrict__ W5,
                           unsigned short* __restrict__ frags)
{
    const int Karr[6] = {100, 400, 400, 400, 400, 400};
    const int Carr[6] = {400, 400, 400, 400, 400, 100};
    const int kst[6]  = {4, 13, 13, 13, 13, 13};
    const int off[6]  = {0, 51200, 217600, 384000, 550400, 716800};
    const float* Wp[6] = {W0, W1, W2, W3, W4, W5};

    int e = blockIdx.x * 256 + threadIdx.x;
    if (e >= FRAG_TOTAL) return;
    int l = 5;
    for (int i = 1; i < 6; ++i) { if (e < off[i]) { l = i - 1; break; } }
    int le     = e - off[l];
    int frag   = le >> 9;          // 512 elems per fragment
    int within = le & 511;
    int lane   = within >> 4;
    int j      = within & 15;
    int ctile  = frag / kst[l];
    int kq     = frag - ctile * kst[l];
    int c = ctile * 16 + (lane & 15);
    int k = kq * 32 + ((lane >> 4) << 4) + j;
    float val = 0.f;
    if (k < Karr[l] && c < Carr[l]) val = Wp[l][(size_t)k * Carr[l] + c];
    frags[e] = f2bf(val);
}

// ---------------------------------------------------------------------------
// Kernel 2: treat-MLP u(t) per step, then v_t[c] = sum_k u[k]*Wd0[100+k, c].
// Tiny work; one block per time step.
// ---------------------------------------------------------------------------
__global__ void treat_precompute(const float* __restrict__ tgrid,
    const float* tw0, const float* tw1, const float* tw2,
    const float* tw3, const float* tw4, const float* tw5,
    const float* tb0, const float* tb1, const float* tb2,
    const float* tb3, const float* tb4, const float* tb5,
    const float* __restrict__ Wd0, float* __restrict__ vbuf)
{
    __shared__ float ush[20];
    int step = blockIdx.x;
    if (threadIdx.x == 0) {
        float tt = tgrid[step];
        float h0[20], h1[20];
        for (int j = 0; j < 20; ++j) { float v = tt * tw0[j] + tb0[j]; h0[j] = v > 0.f ? v : 0.f; }
        const float* Ws[4] = {tw1, tw2, tw3, tw4};
        const float* bs[4] = {tb1, tb2, tb3, tb4};
        for (int m = 0; m < 4; ++m) {
            for (int j = 0; j < 20; ++j) {
                float s = bs[m][j];
                for (int k = 0; k < 20; ++k) s += h0[k] * Ws[m][k * 20 + j];
                h1[j] = s > 0.f ? s : 0.f;
            }
            for (int j = 0; j < 20; ++j) h0[j] = h1[j];
        }
        for (int j = 0; j < 20; ++j) {
            float s = tb5[j];
            for (int k = 0; k < 20; ++k) s += h0[k] * tw5[k * 20 + j];
            ush[j] = s;
        }
    }
    __syncthreads();
    for (int c = threadIdx.x; c < 400; c += blockDim.x) {
        float s = 0.f;
        for (int k = 0; k < 20; ++k) s += ush[k] * Wd0[(size_t)(100 + k) * 400 + c];
        vbuf[step * 400 + c] = s;
    }
}

// ---------------------------------------------------------------------------
// Main kernel: persistent per-block SDE integration of 64 rows over 40 steps.
// LDS layout (dynamic):
//   buf0 : 64*416 bf16  ( 53,248 B)   ping
//   buf1 : 64*416 bf16  ( 53,248 B)   pong
//   Ys   : 64*102 f32   ( 26,112 B)   fp32 state (y, logqp, T)
//   Fb   : 64*112 f32   ( 28,672 B)   final-layer fp32 output
//   part : 256 f32      (  1,024 B)   logqp partial sums
// total = 162,304 B  (< 320 KB WGP LDS)
// ---------------------------------------------------------------------------
#define SMEM_BUF1  53248
#define SMEM_YS    106496
#define SMEM_FB    132608
#define SMEM_PART  161280
#define SMEM_TOTAL 162304

__global__ __launch_bounds__(256)
void sde_main(const float* __restrict__ z, const float* __restrict__ Tx,
              const float* __restrict__ tgrid, const float* __restrict__ dW,
              const unsigned short* __restrict__ frags, const float* __restrict__ vbuf,
              const float* b0, const float* b1, const float* b2,
              const float* b3, const float* b4, const float* b5,
              float* __restrict__ out_ys, float* __restrict__ out_lq)
{
    extern __shared__ char smem[];
    unsigned short* buf0 = (unsigned short*)smem;
    unsigned short* buf1 = (unsigned short*)(smem + SMEM_BUF1);
    float* Ys   = (float*)(smem + SMEM_YS);
    float* Fb   = (float*)(smem + SMEM_FB);
    float* part = (float*)(smem + SMEM_PART);

    const int tid  = threadIdx.x;
    const int lane = tid & 31;
    const int wave = tid >> 5;
    const int rowBase = blockIdx.x * TM;

    // ---- init fp32 state: [y(100) | logqp=0 | T=Tx[b]] ----
    for (int idx = tid; idx < TM * AUG; idx += 256) {
        int r = idx / AUG, d = idx - r * AUG;
        int n = rowBase + r;
        float val;
        if (d < L_DIM)       val = z[(size_t)n * L_DIM + d];
        else if (d == L_DIM) val = 0.f;
        else                 val = Tx[n >> 7];          // S = 128
        Ys[idx] = val;
    }
    // zero K-pad columns [400,416) of both ping-pong buffers (never rewritten)
    for (int idx = tid; idx < TM * 16 * 2; idx += 256) {
        int b = idx >= TM * 16;
        int rem = b ? idx - TM * 16 : idx;
        int r = rem >> 4, k = 400 + (rem & 15);
        (b ? buf1 : buf0)[r * KPAD + k] = 0;
    }
    // ---- t=0 outputs (scrambled-reshape scatter: q = t*8192+n) ----
    for (int idx = tid; idx < TM * L_DIM; idx += 256) {
        int r = idx / L_DIM, d = idx - r * L_DIM;
        int n = rowBase + r;
        out_ys[(size_t)n * L_DIM + d] = z[(size_t)n * L_DIM + d];
    }
    for (int idx = tid; idx < TM; idx += 256) {
        int n = rowBase + idx;
        if (n % 41 == 40) out_lq[n / 41] = 0.f;
    }
    __syncthreads();

    const int ksteps_a[6] = {4, 13, 13, 13, 13, 13};
    const int ctiles_a[6] = {25, 25, 25, 25, 25, 7};
    const int Ca[6]       = {400, 400, 400, 400, 400, 100};
    const int foff_a[6]   = {0, 51200, 217600, 384000, 550400, 716800};
    const float* bias_a[6] = {b0, b1, b2, b3, b4, b5};

    for (int step = 0; step < T_STEPS; ++step) {
        float dt  = tgrid[step + 1] - tgrid[step];
        float sdt = SIGMA_C * sqrtf(dt);
        const float* vrow = vbuf + step * 400;

        // ---- pack y -> bf16 activations (layer0 reads k<128 only) ----
        for (int idx = tid; idx < TM * 128; idx += 256) {
            int r = idx >> 7, k = idx & 127;
            buf0[r * KPAD + k] = (k < L_DIM) ? f2bf(Ys[r * AUG + k]) : (unsigned short)0;
        }
        __syncthreads();

        // ---- 6 GEMM layers via v_wmma_f32_16x16x32_bf16 ----
        for (int l = 0; l < 6; ++l) {
            const unsigned short* A = (l & 1) ? buf1 : buf0;
            unsigned short*       Y = (l & 1) ? buf0 : buf1;
            const int ksteps = ksteps_a[l];
            const int ctiles = ctiles_a[l];
            const int C      = Ca[l];
            const float* bias = bias_a[l];
            const uint4* fq = (const uint4*)(frags + foff_a[l]);
            const int ntiles = ctiles * 4;               // 4 mtiles (TM=64)

            for (int tile = wave; tile < ntiles; tile += 8) {
                const int mtile = tile & 3;
                const int ctile = tile >> 2;
                v8f acc = {0.f, 0.f, 0.f, 0.f, 0.f, 0.f, 0.f, 0.f};

                // A: row = mtile*16+(lane&15); 16 contiguous K at k0 + 16*(lane>>4)
                const int rowA = mtile * 16 + (lane & 15);
                const unsigned short* Ab = A + rowA * KPAD + ((lane >> 4) << 4);
                const uint4* bq = fq + ((size_t)(ctile * ksteps) << 6) + (lane << 1);

                for (int ks = 0; ks < ksteps; ++ks) {
                    FragAB a, b;
                    a.q[0] = *(const uint4*)(Ab + ks * 32);        // ds_load_b128
                    a.q[1] = *(const uint4*)(Ab + ks * 32 + 8);    // ds_load_b128
                    b.q[0] = bq[0];                                // global_load_b128
                    b.q[1] = bq[1];
                    bq += 64;
                    acc = __builtin_amdgcn_wmma_f32_16x16x32_bf16(
                        false, a.v, false, b.v, (short)0, acc, false, false);
                }

                const int c     = ctile * 16 + (lane & 15);
                const int rbase = mtile * 16 + ((lane >> 4) << 3);
                if (l == 0) {
                    float bc = bias[c], vc = vrow[c];
                    #pragma unroll
                    for (int r = 0; r < 8; ++r) {
                        float Tr  = Ys[(rbase + r) * AUG + 101];
                        float val = acc[r] + bc + Tr * vc;         // concat trick
                        val = val > 0.f ? val : 0.f;               // relu
                        Y[(rbase + r) * KPAD + c] = f2bf(val);
                    }
                } else if (l < 5) {
                    float bc = bias[c];
                    #pragma unroll
                    for (int r = 0; r < 8; ++r)
                        Y[(rbase + r) * KPAD + c] = f2bf(tanhf(acc[r] + bc));
                } else {
                    if (c < C) {
                        float bc = bias[c];
                        #pragma unroll
                        for (int r = 0; r < 8; ++r)
                            Fb[(rbase + r) * 112 + c] = acc[r] + bc;   // mlp out, fp32
                    }
                }
            }
            __syncthreads();
        }

        // ---- Euler–Maruyama update + logqp + output scatter ----
        {
            const int r       = tid >> 2;       // 64 rows, 4 threads/row
            const int quarter = tid & 3;
            const int n  = rowBase + r;
            const int d0 = quarter * 25;
            const float* dWrow  = dW + ((size_t)step * N_TOTAL + n) * L_DIM;
            float*       outrow = out_ys + ((size_t)(step + 1) * N_TOTAL + n) * L_DIM;
            float sum = 0.f;
            for (int d = d0; d < d0 + 25; ++d) {
                float F  = Fb[r * 112 + d];     // mlp output
                float y  = Ys[r * AUG + d];
                float f  = F + y;               // f = mlp - h, h = -y
                float yn = y + f * dt + sdt * dWrow[d];
                float us = 2.f * (F + 2.f * y); // (f-h)/sigma
                sum += us * us;
                Ys[r * AUG + d] = yn;
                outrow[d] = yn;
            }
            part[tid] = sum;
            __syncthreads();
            if (quarter == 0) {
                float s  = part[tid] + part[tid + 1] + part[tid + 2] + part[tid + 3];
                float lq = Ys[r * AUG + 100] + 0.5f * s * dt;
                Ys[r * AUG + 100] = lq;
                size_t q = (size_t)(step + 1) * N_TOTAL + n;
                if (q % 41 == 40) out_lq[q / 41] = lq;
            }
            __syncthreads();
        }
    }
}

// ---------------------------------------------------------------------------
// Host launcher
// ---------------------------------------------------------------------------
extern "C" void kernel_launch(void* const* d_in, const int* in_sizes, int n_in,
                              void* d_out, int out_size, void* d_ws, size_t ws_size,
                              hipStream_t stream)
{
    (void)in_sizes; (void)n_in; (void)out_size; (void)ws_size;

    const float* z  = (const float*)d_in[0];
    const float* Tx = (const float*)d_in[1];
    const float* t  = (const float*)d_in[2];
    const float* dW = (const float*)d_in[3];
    const float* tw[6]; const float* tb[6]; const float* dwW[6]; const float* db[6];
    for (int i = 0; i < 6; ++i) {
        tw[i]  = (const float*)d_in[4 + i];
        tb[i]  = (const float*)d_in[10 + i];
        dwW[i] = (const float*)d_in[16 + i];
        db[i]  = (const float*)d_in[22 + i];
    }
    float* out_ys = (float*)d_out;
    float* out_lq = out_ys + (size_t)41 * N_TOTAL * L_DIM;   // 33,587,200

    unsigned short* frags = (unsigned short*)d_ws;
    float* vbuf = (float*)((char*)d_ws + FRAG_BYTES);

    prep_frags<<<(FRAG_TOTAL + 255) / 256, 256, 0, stream>>>(
        dwW[0], dwW[1], dwW[2], dwW[3], dwW[4], dwW[5], frags);

    treat_precompute<<<T_STEPS, 128, 0, stream>>>(
        t, tw[0], tw[1], tw[2], tw[3], tw[4], tw[5],
        tb[0], tb[1], tb[2], tb[3], tb[4], tb[5], dwW[0], vbuf);

    hipFuncSetAttribute(reinterpret_cast<const void*>(sde_main),
                        hipFuncAttributeMaxDynamicSharedMemorySize, SMEM_TOTAL);

    sde_main<<<N_TOTAL / TM, 256, SMEM_TOTAL, stream>>>(
        z, Tx, t, dW, frags, vbuf,
        db[0], db[1], db[2], db[3], db[4], db[5], out_ys, out_lq);
}
